// SupConLoss_1236950581699
// MI455X (gfx1250) — matrix operand order repaired and checked by reading, not compiled, not run on hardware
//
#include <hip/hip_runtime.h>

// ---------------------------------------------------------------------------
// SupConLoss for MI455X (gfx1250, wave32, WMMA f16 16x16x32)
//
// loss = mean_i [ -(T/Tb) * (P_i - C_i*(1/T + log(S_i+eps))) / (C_i+eps) ]
//   S_i = sum_{j!=i} exp(l_ij - 1/T),  l_ij = (x_i . x_j)/T
//   P_i = sum_{j!=i, label match} l_ij
//   C_i = #{j!=i : label match}
// (constant shift 1/T == true row max for unit vectors; exact cancellation)
// ---------------------------------------------------------------------------

typedef __attribute__((ext_vector_type(16))) _Float16 v16h;
typedef __attribute__((ext_vector_type(8)))  _Float16 v8h;
typedef __attribute__((ext_vector_type(8)))  float    v8f;

#define SUPCON_D        128
#define SUPCON_SCALE    (1.0f / 0.3f)             // 1/TEMPERATURE
#define SUPCON_SHIFT    (1.0f / 0.3f)             // constant row max = 1/T
#define SUPCON_RATIO    (0.3f / 0.07f)            // TEMPERATURE / BASE_TEMPERATURE
#define SUPCON_EPS      1e-8f
#define COL_SPLITS      8
#define WAVES_PER_BLOCK 4

// ---------------------------------------------------------------------------
// Kernel 1: f32 [B, V=2, D] -> view-major f16 [N, D]; expand labels to [N].
// contrast row n corresponds to features[n % B, n / B, :].
// ---------------------------------------------------------------------------
__global__ __launch_bounds__(256) void supcon_convert(
    const float* __restrict__ features, const int* __restrict__ labels,
    _Float16* __restrict__ Xh, int* __restrict__ labN, int B, int N) {
  int idx = blockIdx.x * 256 + threadIdx.x;          // over N * D
  if (idx >= N * SUPCON_D) return;
  int n = idx >> 7;            // / 128
  int d = idx & (SUPCON_D - 1);
  int v = (n < B) ? 0 : 1;
  int b = n - v * B;
  float x = features[((size_t)b * 2 + v) * SUPCON_D + d];
  Xh[(size_t)n * SUPCON_D + d] = (_Float16)x;
  if (d == 0) labN[n] = labels[b];
}

// ---------------------------------------------------------------------------
// Kernel 2: streaming logits tiles via v_wmma_f32_16x16x32_f16.
// One wave handles a 16-row tile over one column split; B fragments are
// double-buffered in registers (loop unrolled x2) so the next tile's load
// clause overlaps the current tile's WMMA chain.
// C-fragment element (reg r, lane l) = tile[M = r + 8*(l>>4), Ncol = l&15].
// A frag (16-bit 16x32): lane l: row=rb+(l&15); V0..7 <- K kb..kb+15,
//                        kb = 16*(l>>4).
// B frag (16-bit 32x16): lane l: col=cb+(l&15); V0..7 <- K kh..kh+15,
//                        kh = 16*(l>>4).  (B[k,n] = X[n,k])
// ---------------------------------------------------------------------------
__global__ __launch_bounds__(128) void supcon_main(
    const _Float16* __restrict__ Xh, const int* __restrict__ labN,
    float* __restrict__ wsS, float* __restrict__ wsP, float* __restrict__ wsC,
    int N, int colsPerSplit) {
  const int lane = threadIdx.x & 31;
  const int wave = threadIdx.x >> 5;
  const int rb = (blockIdx.x * WAVES_PER_BLOCK + wave) * 16;  // row tile base
  const int split = blockIdx.y;
  const int cb0 = split * colsPerSplit;

  const int lm = lane & 15;
  const int lh = lane >> 4;

  // ---- loop-invariant A fragments (4 K-steps of 32) ----
  const _Float16* __restrict__ rowPtr = Xh + (size_t)(rb + lm) * SUPCON_D;
  v16h afrag[4];
#pragma unroll
  for (int kk = 0; kk < 4; ++kk) {
    const _Float16* p = rowPtr + kk * 32 + lh * 16;
    *(v8h*)&afrag[kk]       = *(const v8h*)(p);        // K kb..kb+7    -> V0..V3
    *((v8h*)&afrag[kk] + 1) = *(const v8h*)(p + 8);    // K kb+8..+15   -> V4..V7
  }

  // ---- per-register row ids and labels (loop invariant) ----
  int rg[8], rlab[8];
#pragma unroll
  for (int r = 0; r < 8; ++r) {
    rg[r] = rb + r + lh * 8;
    rlab[r] = labN[rg[r]];
  }

  float Sacc[8], Pacc[8], Cacc[8];
#pragma unroll
  for (int r = 0; r < 8; ++r) { Sacc[r] = 0.f; Pacc[r] = 0.f; Cacc[r] = 0.f; }

  const int nTiles = colsPerSplit >> 4;   // multiple of 2 (N/COL_SPLITS/16)

  // load one B fragment set (4 K-steps) for column tile `tile`
  auto loadB = [&](int tile, v16h bf[4]) {
    const _Float16* __restrict__ colPtr =
        Xh + (size_t)(cb0 + tile * 16 + lm) * SUPCON_D + lh * 16;
#pragma unroll
    for (int kk = 0; kk < 4; ++kk) {
      const _Float16* p = colPtr + kk * 32;
      *(v8h*)&bf[kk]       = *(const v8h*)(p);         // K kh..kh+7   -> V0..V3
      *((v8h*)&bf[kk] + 1) = *(const v8h*)(p + 8);     // K kh+8..+15  -> V4..V7
    }
  };

  // 4-step WMMA chain + masked softmax-sum epilogue for column tile `tile`
  auto compute = [&](int tile, v16h bf[4]) {
    const int cg = cb0 + tile * 16 + lm;       // this lane's global column
    const int clab = labN[cg];
    v8f acc = {0.f, 0.f, 0.f, 0.f, 0.f, 0.f, 0.f, 0.f};
#pragma unroll
    for (int kk = 0; kk < 4; ++kk) {
      acc = __builtin_amdgcn_wmma_f32_16x16x32_f16(
          /*neg_a=*/false, afrag[kk], /*neg_b=*/false, bf[kk],
          /*c_mod=*/(short)0, acc, /*reuse_a=*/false, /*reuse_b=*/false);
    }
#pragma unroll
    for (int r = 0; r < 8; ++r) {
      float c = acc[r];                        // raw dot product
      bool diag  = (rg[r] == cg);
      bool match = (!diag) && (rlab[r] == clab);
      float e = diag ? 0.0f : __expf(c * SUPCON_SCALE - SUPCON_SHIFT);
      Sacc[r] += e;
      Pacc[r] += match ? c : 0.0f;             // raw; scaled at write-out
      Cacc[r] += match ? 1.0f : 0.0f;
    }
  };

  // ---- software-pipelined main loop (double-buffered B fragments) ----
  v16h b0[4], b1[4];
  loadB(0, b0);
  for (int ct = 0; ct < nTiles; ct += 2) {
    loadB(ct + 1, b1);                         // prefetch while computing b0
    compute(ct, b0);
    int tn = (ct + 2 < nTiles) ? ct + 2 : (nTiles - 1);  // clamped (dead last)
    loadB(tn, b0);                             // prefetch while computing b1
    compute(ct + 1, b1);
  }

  // ---- reduce across the 16 lanes of each half (columns of the tile) ----
#pragma unroll
  for (int off = 1; off < 16; off <<= 1) {
#pragma unroll
    for (int r = 0; r < 8; ++r) {
      Sacc[r] += __shfl_xor(Sacc[r], off, 32);
      Pacc[r] += __shfl_xor(Pacc[r], off, 32);
      Cacc[r] += __shfl_xor(Cacc[r], off, 32);
    }
  }

  if (lm == 0) {
#pragma unroll
    for (int r = 0; r < 8; ++r) {
      size_t idx = (size_t)split * N + rg[r];
      wsS[idx] = Sacc[r];
      wsP[idx] = Pacc[r] * SUPCON_SCALE;       // apply 1/T once here
      wsC[idx] = Cacc[r];
    }
  }
}

// ---------------------------------------------------------------------------
// Kernel 3: combine column splits, per-row loss, deterministic mean.
// ---------------------------------------------------------------------------
__global__ __launch_bounds__(256) void supcon_finish(
    const float* __restrict__ wsS, const float* __restrict__ wsP,
    const float* __restrict__ wsC, float* __restrict__ out, int N) {
  __shared__ float red[256];
  float local = 0.f;
  for (int row = threadIdx.x; row < N; row += 256) {
    float S = 0.f, P = 0.f, C = 0.f;
    for (int s = 0; s < COL_SPLITS; ++s) {
      size_t idx = (size_t)s * N + row;
      S += wsS[idx]; P += wsP[idx]; C += wsC[idx];
    }
    float mlpp = (P - C * (SUPCON_SHIFT + __logf(S + SUPCON_EPS))) / (C + SUPCON_EPS);
    local += -SUPCON_RATIO * mlpp;
  }
  red[threadIdx.x] = local;
  __syncthreads();
  for (int off = 128; off > 0; off >>= 1) {
    if (threadIdx.x < off) red[threadIdx.x] += red[threadIdx.x + off];
    __syncthreads();
  }
  if (threadIdx.x == 0) out[0] = red[0] / (float)N;
}

// ---------------------------------------------------------------------------
extern "C" void kernel_launch(void* const* d_in, const int* in_sizes, int n_in,
                              void* d_out, int out_size, void* d_ws, size_t ws_size,
                              hipStream_t stream) {
  const float* features = (const float*)d_in[0];   // [B, 2, 128] f32
  const int*   labels   = (const int*)d_in[1];     // [B]
  const int B = in_sizes[1];
  const int N = 2 * B;                             // 8192

  // workspace partition (256-byte aligned slices)
  char* ws = (char*)d_ws;
  size_t off = 0;
  _Float16* Xh = (_Float16*)(ws + off);
  off += (size_t)N * SUPCON_D * sizeof(_Float16);
  off = (off + 255) & ~(size_t)255;
  int* labN = (int*)(ws + off);
  off += (size_t)N * sizeof(int);
  off = (off + 255) & ~(size_t)255;
  float* wsS = (float*)(ws + off);
  off += (size_t)COL_SPLITS * N * sizeof(float);
  off = (off + 255) & ~(size_t)255;
  float* wsP = (float*)(ws + off);
  off += (size_t)COL_SPLITS * N * sizeof(float);
  off = (off + 255) & ~(size_t)255;
  float* wsC = (float*)(ws + off);

  // 1) convert + label expand
  int totalElems = N * SUPCON_D;
  supcon_convert<<<(totalElems + 255) / 256, 256, 0, stream>>>(
      features, labels, Xh, labN, B, N);

  // 2) WMMA streaming logits reductions
  dim3 grid(N / (16 * WAVES_PER_BLOCK), COL_SPLITS);
  supcon_main<<<grid, 32 * WAVES_PER_BLOCK, 0, stream>>>(
      Xh, labN, wsS, wsP, wsC, N, N / COL_SPLITS);

  // 3) per-row loss + deterministic mean
  supcon_finish<<<1, 256, 0, stream>>>(wsS, wsP, wsC, (float*)d_out, N);
}
